// Conditioned_Mlp_34952443854980
// MI455X (gfx1250) — matrix operand order repaired and checked by reading, not compiled, not run on hardware
//
#include <hip/hip_runtime.h>
#include <stdint.h>

#define N_TOK 16384
#define DDIM  1024
#define EEXP  4
#define HDIM  4096

typedef __attribute__((ext_vector_type(16))) __bf16 v16bf;
typedef __attribute__((ext_vector_type(8)))  float  v8f;
typedef __attribute__((ext_vector_type(4)))  int    v4i;

#if defined(__gfx1250__) && __has_builtin(__builtin_amdgcn_global_load_async_to_lds_b128) && \
    __has_builtin(__builtin_amdgcn_s_wait_asynccnt)
#define USE_ASYNC 1
#else
#define USE_ASYNC 0
#endif

#if USE_ASYNC
#define ASG __attribute__((address_space(1)))
#define ASL __attribute__((address_space(3)))
__device__ __forceinline__ ASG v4i* gcast(const void* p) { return (ASG v4i*)p; }
__device__ __forceinline__ ASL v4i* lcast(void* p)       { return (ASL v4i*)p; }
#endif

__device__ __forceinline__ unsigned short f2bf(float f) {
    unsigned int u = __float_as_uint(f);
    u += 0x7FFFu + ((u >> 16) & 1u);            // round-to-nearest-even
    return (unsigned short)(u >> 16);
}

// ---------------------------------------------------------------------------
// Gate: logits = concat(q,k) @ Wg + bg ; softmax over 4 experts.
// One wave32 per token, lanes stride over D, shfl_xor reduction.
// ---------------------------------------------------------------------------
__global__ void gate_kernel(const float* __restrict__ q, const float* __restrict__ k,
                            const float* __restrict__ Wg, const float* __restrict__ bg,
                            float* __restrict__ gate) {
    const int tok  = (blockIdx.x * blockDim.x + threadIdx.x) >> 5;
    const int lane = threadIdx.x & 31;
    const float* qr = q + (size_t)tok * DDIM;
    const float* kr = k + (size_t)tok * DDIM;
    float a0 = 0.f, a1 = 0.f, a2 = 0.f, a3 = 0.f;
    for (int d = lane; d < DDIM; d += 32) {
        float qv = qr[d], kv = kr[d];
        float4 wq = *(const float4*)(Wg + (size_t)d * 4);
        float4 wk = *(const float4*)(Wg + (size_t)(DDIM + d) * 4);
        a0 += qv * wq.x + kv * wk.x;
        a1 += qv * wq.y + kv * wk.y;
        a2 += qv * wq.z + kv * wk.z;
        a3 += qv * wq.w + kv * wk.w;
    }
    #pragma unroll
    for (int off = 16; off > 0; off >>= 1) {
        a0 += __shfl_xor(a0, off, 32);
        a1 += __shfl_xor(a1, off, 32);
        a2 += __shfl_xor(a2, off, 32);
        a3 += __shfl_xor(a3, off, 32);
    }
    if (lane == 0) {
        a0 += bg[0]; a1 += bg[1]; a2 += bg[2]; a3 += bg[3];   // TEMP == 1.0
        float m  = fmaxf(fmaxf(a0, a1), fmaxf(a2, a3));
        float e0 = __expf(a0 - m), e1 = __expf(a1 - m);
        float e2 = __expf(a2 - m), e3 = __expf(a3 - m);
        float inv = 1.f / (e0 + e1 + e2 + e3);
        float4 g; g.x = e0 * inv; g.y = e1 * inv; g.z = e2 * inv; g.w = e3 * inv;
        *(float4*)(gate + (size_t)tok * 4) = g;
    }
}

// ---------------------------------------------------------------------------
// Elementwise f32 -> bf16 convert (q).
// ---------------------------------------------------------------------------
__global__ void cvt_kernel(const float* __restrict__ in, unsigned short* __restrict__ out) {
    const size_t i = (size_t)blockIdx.x * blockDim.x + threadIdx.x;   // over N*D/4
    float4 v = *(const float4*)(in + i * 4);
    unsigned int lo = (unsigned)f2bf(v.x) | ((unsigned)f2bf(v.y) << 16);
    unsigned int hi = (unsigned)f2bf(v.z) | ((unsigned)f2bf(v.w) << 16);
    uint2 pk; pk.x = lo; pk.y = hi;
    *(uint2*)(out + i * 4) = pk;
}

// ---------------------------------------------------------------------------
// Tiled transpose + convert: in [E][R][C] f32 -> out [E][C][R] bf16.
// ---------------------------------------------------------------------------
__global__ void tr_cvt_kernel(const float* __restrict__ in, unsigned short* __restrict__ out,
                              int R, int C) {
    __shared__ unsigned short tile[32][33];
    const float*         src = in  + (size_t)blockIdx.z * R * C;
    unsigned short*      dst = out + (size_t)blockIdx.z * R * C;
    const int c0 = blockIdx.x * 32, r0 = blockIdx.y * 32;
    const int tx = threadIdx.x, ty = threadIdx.y;          // block (32,8)
    #pragma unroll
    for (int i = 0; i < 32; i += 8)
        tile[ty + i][tx] = f2bf(src[(size_t)(r0 + ty + i) * C + (c0 + tx)]);
    __syncthreads();
    #pragma unroll
    for (int i = 0; i < 32; i += 8)
        dst[(size_t)(c0 + ty + i) * R + (r0 + tx)] = tile[tx][ty + i];
}

// ---------------------------------------------------------------------------
// out[n,d] = sum_e gate[n,e] * b2[e,d]   (bias term of the gated combine)
// ---------------------------------------------------------------------------
__global__ void init_out_kernel(const float* __restrict__ gate, const float* __restrict__ b2,
                                float* __restrict__ out) {
    const size_t i = (size_t)blockIdx.x * blockDim.x + threadIdx.x;   // over N*D
    const int n = (int)(i >> 10);
    const int d = (int)(i & (DDIM - 1));
    float4 g = *(const float4*)(gate + (size_t)n * 4);
    out[i] = g.x * b2[d] + g.y * b2[DDIM + d] + g.z * b2[2 * DDIM + d] + g.w * b2[3 * DDIM + d];
}

// ---------------------------------------------------------------------------
// bf16 WMMA GEMM, 128x128 block tile, 8 waves, wave tile 32x64 (2x4 wmma),
// K-step 32, double-buffered LDS, A:[MxK] row-major, Bt:[NxK] row-major (B^T).
// Staging: async LDS DMA (GLOBAL_LOAD_ASYNC_TO_LDS_B128 / ASYNCcnt) when the
// toolchain exposes it, else global->reg->ds_store.
// MODE 0: Hout = relu(A@B + bias)  stored bf16
// MODE 1: Out += gate[:,e] * (A@B) stored f32 (read-modify-write)
// ---------------------------------------------------------------------------
template <int MODE>
__global__ __launch_bounds__(256) void gemm_kernel(
    const unsigned short* __restrict__ A,
    const unsigned short* __restrict__ Bt,
    int K,                                   // reduction length
    int Nld,                                 // output row length (ld)
    const float* __restrict__ bias,          // MODE 0: b1_e [Nld]
    const float* __restrict__ gate,          // MODE 1: [N_TOK][4]
    int e,
    unsigned short* __restrict__ Hout,       // MODE 0
    float* __restrict__ Out)                 // MODE 1
{
    __shared__ unsigned short As[2][128 * 40];
    __shared__ unsigned short Bs[2][128 * 40];

    const int tid  = threadIdx.x;
    const int lane = tid & 31;
    const int wave = tid >> 5;
    const int wr   = wave >> 1;              // 0..3 : 32-row band
    const int wc   = wave & 1;               // 0..1 : 64-col band
    const int m0   = blockIdx.y * 128;
    const int n0   = blockIdx.x * 128;

    // staging: 512 x 16B chunks per tile, 2 per thread
    const int sr = tid >> 2;                 // 0..63
    const int sq = tid & 3;                  // 16B quarter of a 64B row
    const size_t aofs0 = (size_t)(m0 + sr)      * K + sq * 8;
    const size_t aofs1 = (size_t)(m0 + 64 + sr) * K + sq * 8;
    const size_t bofs0 = (size_t)(n0 + sr)      * K + sq * 8;
    const size_t bofs1 = (size_t)(n0 + 64 + sr) * K + sq * 8;
    const int ldsi0 = sr * 40 + sq * 8;
    const int ldsi1 = (64 + sr) * 40 + sq * 8;

    v8f acc[2][4] = {};

    const int nk = K >> 5;
    const int ml = lane & 15;
    const int hf = lane >> 4;

#if USE_ASYNC
    // ---- async DMA staging path (ASYNCcnt) ----
    {
        __builtin_amdgcn_global_load_async_to_lds_b128(gcast(A  + aofs0), lcast(&As[0][ldsi0]), 0, 0);
        __builtin_amdgcn_global_load_async_to_lds_b128(gcast(A  + aofs1), lcast(&As[0][ldsi1]), 0, 0);
        __builtin_amdgcn_global_load_async_to_lds_b128(gcast(Bt + bofs0), lcast(&Bs[0][ldsi0]), 0, 0);
        __builtin_amdgcn_global_load_async_to_lds_b128(gcast(Bt + bofs1), lcast(&Bs[0][ldsi1]), 0, 0);
    }
    __builtin_amdgcn_s_wait_asynccnt(0);
    __syncthreads();
#else
    uint4 pa0 = *(const uint4*)(A  + aofs0);
    uint4 pa1 = *(const uint4*)(A  + aofs1);
    uint4 pb0 = *(const uint4*)(Bt + bofs0);
    uint4 pb1 = *(const uint4*)(Bt + bofs1);
    *(uint4*)&As[0][ldsi0] = pa0;
    *(uint4*)&As[0][ldsi1] = pa1;
    *(uint4*)&Bs[0][ldsi0] = pb0;
    *(uint4*)&Bs[0][ldsi1] = pb1;
    __syncthreads();
#endif

    for (int ks = 0; ks < nk; ++ks) {
        const int  buf  = ks & 1;
        const bool more = (ks + 1) < nk;
        if (more) {
            const int k0 = (ks + 1) << 5;
#if USE_ASYNC
            const int nb = buf ^ 1;
            __builtin_amdgcn_global_load_async_to_lds_b128(gcast(A  + aofs0 + k0), lcast(&As[nb][ldsi0]), 0, 0);
            __builtin_amdgcn_global_load_async_to_lds_b128(gcast(A  + aofs1 + k0), lcast(&As[nb][ldsi1]), 0, 0);
            __builtin_amdgcn_global_load_async_to_lds_b128(gcast(Bt + bofs0 + k0), lcast(&Bs[nb][ldsi0]), 0, 0);
            __builtin_amdgcn_global_load_async_to_lds_b128(gcast(Bt + bofs1 + k0), lcast(&Bs[nb][ldsi1]), 0, 0);
#else
            pa0 = *(const uint4*)(A  + aofs0 + k0);
            pa1 = *(const uint4*)(A  + aofs1 + k0);
            pb0 = *(const uint4*)(Bt + bofs0 + k0);
            pb1 = *(const uint4*)(Bt + bofs1 + k0);
#endif
            if ((ks + 2) < nk) {               // L2 prefetch two steps ahead
                const int k2 = (ks + 2) << 5;
                __builtin_prefetch(A  + aofs0 + k2, 0, 1);
                __builtin_prefetch(Bt + bofs0 + k2, 0, 1);
            }
        }

        // --- fragments from LDS (documented 16-bit WMMA A/B lane layouts) ---
        const unsigned short* ab = &As[buf][0];
        const unsigned short* bb = &Bs[buf][0];
        v16bf af[2], bfr[4];
        #pragma unroll
        for (int ti = 0; ti < 2; ++ti) {
            const unsigned short* p = ab + (wr * 32 + ti * 16 + ml) * 40;
            *(uint4*)&af[ti]       = *(const uint4*)(p + hf * 8);        // K chunk 0
            *((uint4*)&af[ti] + 1) = *(const uint4*)(p + 16 + hf * 8);   // K chunk 1
        }
        #pragma unroll
        for (int tj = 0; tj < 4; ++tj) {
            const unsigned short* p = bb + (wc * 64 + tj * 16 + ml) * 40;
            *(uint4*)&bfr[tj]       = *(const uint4*)(p + hf * 8);
            *((uint4*)&bfr[tj] + 1) = *(const uint4*)(p + 16 + hf * 8);
        }
        #pragma unroll
        for (int ti = 0; ti < 2; ++ti)
            #pragma unroll
            for (int tj = 0; tj < 4; ++tj)
                acc[ti][tj] = __builtin_amdgcn_wmma_f32_16x16x32_bf16(
                    false, af[ti], false, bfr[tj], (short)0, acc[ti][tj], false, false);

#if USE_ASYNC
        __builtin_amdgcn_s_wait_asynccnt(0);   // next buffer resident in LDS
#else
        if (more) {
            const int nb = buf ^ 1;
            *(uint4*)&As[nb][ldsi0] = pa0;
            *(uint4*)&As[nb][ldsi1] = pa1;
            *(uint4*)&Bs[nb][ldsi0] = pb0;
            *(uint4*)&Bs[nb][ldsi1] = pb1;
        }
#endif
        __syncthreads();
    }

    // --- epilogue: C layout = (vgpr v, lane) -> M = v + 8*hf, N = lane&15 ---
    #pragma unroll
    for (int ti = 0; ti < 2; ++ti) {
        const int mbase = m0 + wr * 32 + ti * 16 + hf * 8;
        float gv[8];
        if (MODE == 1) {
            #pragma unroll
            for (int v = 0; v < 8; ++v)
                gv[v] = gate[(size_t)(mbase + v) * 4 + e];
        }
        #pragma unroll
        for (int tj = 0; tj < 4; ++tj) {
            const int n = n0 + wc * 64 + tj * 16 + ml;
            v8f c = acc[ti][tj];
            if (MODE == 0) {
                const float b = bias[n];
                #pragma unroll
                for (int v = 0; v < 8; ++v) {
                    float x = fmaxf(c[v] + b, 0.f);
                    Hout[(size_t)(mbase + v) * Nld + n] = f2bf(x);
                }
            } else {
                #pragma unroll
                for (int v = 0; v < 8; ++v) {
                    const size_t o = (size_t)(mbase + v) * Nld + n;
                    Out[o] = fmaf(c[v], gv[v], Out[o]);
                }
            }
        }
    }
}

// ---------------------------------------------------------------------------
extern "C" void kernel_launch(void* const* d_in, const int* in_sizes, int n_in,
                              void* d_out, int out_size, void* d_ws, size_t ws_size,
                              hipStream_t stream) {
    const float* q  = (const float*)d_in[0];
    const float* k  = (const float*)d_in[1];
    const float* W1 = (const float*)d_in[2];
    const float* b1 = (const float*)d_in[3];
    const float* W2 = (const float*)d_in[4];
    const float* b2 = (const float*)d_in[5];
    const float* Wg = (const float*)d_in[6];
    const float* bg = (const float*)d_in[7];
    float* out = (float*)d_out;

    // workspace carve (256B aligned)
    char* ws = (char*)d_ws;
    size_t off = 0;
    auto carve = [&](size_t bytes) -> char* {
        char* p = ws + off;
        off = (off + bytes + 255) & ~(size_t)255;
        return p;
    };
    float*          gateb = (float*)         carve((size_t)N_TOK * 4 * sizeof(float));
    unsigned short* qb    = (unsigned short*)carve((size_t)N_TOK * DDIM * 2);
    unsigned short* w1t   = (unsigned short*)carve((size_t)EEXP * HDIM * DDIM * 2);  // [E][H][D]
    unsigned short* w2t   = (unsigned short*)carve((size_t)EEXP * DDIM * HDIM * 2);  // [E][D][H]
    unsigned short* hbuf  = (unsigned short*)carve((size_t)N_TOK * HDIM * 2);        // [N][H], reused per expert

    // 1) gate
    gate_kernel<<<N_TOK / 8, 256, 0, stream>>>(q, k, Wg, bg, gateb);
    // 2) q -> bf16
    cvt_kernel<<<(N_TOK * DDIM / 4) / 256, 256, 0, stream>>>(q, qb);
    // 3) W1 [E][D][H] -> w1t [E][H][D] bf16
    tr_cvt_kernel<<<dim3(HDIM / 32, DDIM / 32, EEXP), dim3(32, 8), 0, stream>>>(W1, w1t, DDIM, HDIM);
    // 4) W2 [E][H][D] -> w2t [E][D][H] bf16
    tr_cvt_kernel<<<dim3(DDIM / 32, HDIM / 32, EEXP), dim3(32, 8), 0, stream>>>(W2, w2t, HDIM, DDIM);
    // 5) out = sum_e gate_e * b2_e
    init_out_kernel<<<(N_TOK * DDIM) / 256, 256, 0, stream>>>(gateb, b2, out);

    // 6) per expert: h = relu(qb @ W1_e + b1_e); out += gate_e * (h @ W2_e)
    for (int e = 0; e < EEXP; ++e) {
        gemm_kernel<0><<<dim3(HDIM / 128, N_TOK / 128), 256, 0, stream>>>(
            qb, w1t + (size_t)e * HDIM * DDIM, DDIM, HDIM,
            b1 + (size_t)e * HDIM, nullptr, e, hbuf, nullptr);
        gemm_kernel<1><<<dim3(DDIM / 128, N_TOK / 128), 256, 0, stream>>>(
            hbuf, w2t + (size_t)e * DDIM * HDIM, HDIM, DDIM,
            nullptr, gateb, e, nullptr, out);
    }
}
